// SenriAttention_20478404067410
// MI455X (gfx1250) — compile-verified, matrix operand order checked by reading, not run on hardware
//
#include <hip/hip_runtime.h>
#include <hip/hip_bf16.h>

// ---------------- problem constants ----------------
#define B_   2
#define S_   4096
#define H_   2048
#define D_   128
#define R_   (B_*S_)        // 8192 token rows
#define L_   128            // linear-attention chunk length
#define NCH  (R_/L_)        // 64 chunks total (32 per batch, batch-aligned)
#define EPSF 1e-6f

typedef __attribute__((ext_vector_type(16))) __bf16 v16bf;
typedef __attribute__((ext_vector_type(8)))  __bf16 v8bf;
typedef __attribute__((ext_vector_type(8)))  float  v8f;
typedef __attribute__((ext_vector_type(4)))  unsigned int v4u;
typedef __attribute__((ext_vector_type(8)))  int    v8i;
typedef __attribute__((ext_vector_type(4)))  int    v4i;

#if __has_builtin(__builtin_amdgcn_tensor_load_to_lds)
#define HAVE_TDM 1
#else
#define HAVE_TDM 0
#endif

// ---------------- helpers ----------------
__device__ __forceinline__ float b2f(__bf16 x) { return (float)x; }
__device__ __forceinline__ __bf16 f2b(float x) { return (__bf16)x; }

__device__ __forceinline__ float sigma_elu1(float x) {
    return x > 0.f ? x + 1.f : __expf(x);   // elu(x)+1
}

// Load a 16x32 bf16 WMMA fragment (A or B) from a row-major [16 x K] slice.
// ISA 16-bit layout: lanes 0-15 hold K {0..7,16..23}, lanes 16-31 hold K {8..15,24..31}.
__device__ __forceinline__ v16bf load_frag(const __bf16* base, int ldk) {
    int lane = threadIdx.x & 31;
    int hf   = lane >> 4;
    int row  = lane & 15;
    const __bf16* p = base + (size_t)row * ldk + hf * 8;
    v8bf c0 = *(const v8bf*)(p);
    v8bf c1 = *(const v8bf*)(p + 16);
    v16bf r;
#pragma unroll
    for (int i = 0; i < 8; ++i) { r[i] = c0[i]; r[i + 8] = c1[i]; }
    return r;
}

__device__ __forceinline__ v8f wmma_bf16(v16bf a, v16bf b, v8f c) {
    return __builtin_amdgcn_wmma_f32_16x16x32_bf16(
        false, a, false, b, (short)0, c, false, false);
}

// C/D 16x16 f32 layout: VGPR i -> M = i + 8*(lane>>4), N = lane&15
#define CD_INDEX(lane, i, m, n) do { (m) = (i) + 8 * ((lane) >> 4); (n) = (lane) & 15; } while (0)

// Register-blocked 32Mx64N GEMM body: 2 A-frags x 4 B-frags -> 8 WMMA per K-step.
__device__ __forceinline__ void gemm_32x64(const __bf16* A, int lda,
                                           const __bf16* Bw, int ldb,
                                           int ktot, v8f acc[2][4]) {
    for (int k = 0; k < ktot; k += 32) {
        __builtin_prefetch(A + k + 512, 0, 0);
        __builtin_prefetch(Bw + k + 512, 0, 0);
        v16bf a0 = load_frag(A + k, lda);
        v16bf a1 = load_frag(A + (size_t)16 * lda + k, lda);
        v16bf b0 = load_frag(Bw + k, ldb);
        v16bf b1 = load_frag(Bw + (size_t)16 * ldb + k, ldb);
        v16bf b2 = load_frag(Bw + (size_t)32 * ldb + k, ldb);
        v16bf b3 = load_frag(Bw + (size_t)48 * ldb + k, ldb);
        acc[0][0] = wmma_bf16(a0, b0, acc[0][0]);
        acc[0][1] = wmma_bf16(a0, b1, acc[0][1]);
        acc[0][2] = wmma_bf16(a0, b2, acc[0][2]);
        acc[0][3] = wmma_bf16(a0, b3, acc[0][3]);
        acc[1][0] = wmma_bf16(a1, b0, acc[1][0]);
        acc[1][1] = wmma_bf16(a1, b1, acc[1][1]);
        acc[1][2] = wmma_bf16(a1, b2, acc[1][2]);
        acc[1][3] = wmma_bf16(a1, b3, acc[1][3]);
    }
}

// ---------------- kernels ----------------

__global__ void k_cvt_bf16(const float* __restrict__ src, __bf16* __restrict__ dst, int n) {
    for (int i = blockIdx.x * blockDim.x + threadIdx.x; i < n; i += gridDim.x * blockDim.x)
        dst[i] = f2b(src[i]);
}

__global__ void k_transpose_Mt(const float* __restrict__ M, __bf16* __restrict__ Mt) {
    int i = blockIdx.x * blockDim.x + threadIdx.x;
    if (i >= D_ * D_) return;
    int d = i / D_, e = i % D_;
    Mt[e * D_ + d] = f2b(M[d * D_ + e]);
}

// QKV projection: 32Mx64N tile per wave; fused sigma for q,k; transposed chunk copies.
__global__ void k_gemm_qkv(const __bf16* __restrict__ hb, const __bf16* __restrict__ wb,
                           __bf16* __restrict__ qsb, __bf16* __restrict__ ksb,
                           __bf16* __restrict__ pkT, float* __restrict__ vf,
                           __bf16* __restrict__ vT) {
    int m0 = blockIdx.x * 32;
    int nt = blockIdx.y;                 // 0..5
    int which = nt >> 1;                 // 0=q 1=k 2=v
    int n0 = (nt & 1) * 64;
    v8f acc[2][4] = {};
    gemm_32x64(hb + (size_t)m0 * H_, H_,
               wb + ((size_t)which * D_ + n0) * H_, H_, H_, acc);
    int lane = threadIdx.x & 31;
#pragma unroll
    for (int mi = 0; mi < 2; ++mi)
#pragma unroll
        for (int ni = 0; ni < 4; ++ni)
#pragma unroll
            for (int i = 0; i < 8; ++i) {
                int m, n; CD_INDEX(lane, i, m, n);
                int r   = m0 + 16 * mi + m;
                int col = n0 + 16 * ni + n;
                float val = acc[mi][ni][i];
                if (which == 0) {
                    qsb[(size_t)r * D_ + col] = f2b(sigma_elu1(val));
                } else if (which == 1) {
                    float s = sigma_elu1(val);
                    ksb[(size_t)r * D_ + col] = f2b(s);
                    int ch = r / L_, l = r % L_;
                    pkT[((size_t)ch * D_ + col) * L_ + l] = f2b(s);
                } else {
                    vf[(size_t)r * D_ + col] = val;
                    int ch = r / L_, l = r % L_;
                    vT[((size_t)ch * D_ + col) * L_ + l] = f2b(val);
                }
            }
}

__global__ void k_row_dots(const __bf16* __restrict__ qsb, const __bf16* __restrict__ ksb,
                           const float* __restrict__ z,
                           float* __restrict__ qzden, float* __restrict__ kzden) {
    int r = blockIdx.x * blockDim.x + threadIdx.x;
    if (r >= R_) return;
    float aq = 0.f, ak = 0.f;
    for (int d = 0; d < D_; ++d) {
        float zd = z[d];
        aq += b2f(qsb[(size_t)r * D_ + d]) * zd;
        ak += b2f(ksb[(size_t)r * D_ + d]) * zd;
    }
    qzden[r] = aq + EPSF;
    kzden[r] = ak + EPSF;
}

// memo = (sigma(q) @ M) / qzden, 32x64 tiles
__global__ void k_retrieve(const __bf16* __restrict__ qsb, const __bf16* __restrict__ Mt,
                           const float* __restrict__ qzden, float* __restrict__ memo) {
    int m0 = blockIdx.x * 32, n0 = blockIdx.y * 64;
    v8f acc[2][4] = {};
    gemm_32x64(qsb + (size_t)m0 * D_, D_, Mt + (size_t)n0 * D_, D_, D_, acc);
    int lane = threadIdx.x & 31;
#pragma unroll
    for (int mi = 0; mi < 2; ++mi)
#pragma unroll
        for (int ni = 0; ni < 4; ++ni)
#pragma unroll
            for (int i = 0; i < 8; ++i) {
                int m, n; CD_INDEX(lane, i, m, n);
                int r = m0 + 16 * mi + m;
                memo[(size_t)r * D_ + n0 + 16 * ni + n] = acc[mi][ni][i] / qzden[r];
            }
}

// v_ret; write diffT[e][r] = bf16(v - v_ret)
__global__ void k_vret_diff(const __bf16* __restrict__ ksb, const __bf16* __restrict__ Mt,
                            const float* __restrict__ kzden, const float* __restrict__ vf,
                            __bf16* __restrict__ diffT) {
    int m0 = blockIdx.x * 32, n0 = blockIdx.y * 64;
    v8f acc[2][4] = {};
    gemm_32x64(ksb + (size_t)m0 * D_, D_, Mt + (size_t)n0 * D_, D_, D_, acc);
    int lane = threadIdx.x & 31;
#pragma unroll
    for (int mi = 0; mi < 2; ++mi)
#pragma unroll
        for (int ni = 0; ni < 4; ++ni)
#pragma unroll
            for (int i = 0; i < 8; ++i) {
                int m, n; CD_INDEX(lane, i, m, n);
                int r = m0 + 16 * mi + m, e = n0 + 16 * ni + n;
                float vret = acc[mi][ni][i] / kzden[r];
                diffT[(size_t)e * R_ + r] = f2b(vf[(size_t)r * D_ + e] - vret);
            }
}

// per-chunk KtV[ch][d][e] = sum_l sigma(k)[l,d]*v[l,e]; 32x64 tiles
__global__ void k_chunk_kv(const __bf16* __restrict__ pkT, const __bf16* __restrict__ vT,
                           float* __restrict__ KtV) {
    int ch = blockIdx.z;
    int d0 = blockIdx.x * 32, e0 = blockIdx.y * 64;
    v8f acc[2][4] = {};
    gemm_32x64(pkT + ((size_t)ch * D_ + d0) * L_, L_,
               vT  + ((size_t)ch * D_ + e0) * L_, L_, L_, acc);
    int lane = threadIdx.x & 31;
#pragma unroll
    for (int mi = 0; mi < 2; ++mi)
#pragma unroll
        for (int ni = 0; ni < 4; ++ni)
#pragma unroll
            for (int i = 0; i < 8; ++i) {
                int m, n; CD_INDEX(lane, i, m, n);
                KtV[(size_t)ch * D_ * D_ + (size_t)(d0 + 16 * mi + m) * D_ +
                    e0 + 16 * ni + n] = acc[mi][ni][i];
            }
}

__global__ void k_chunk_ksum(const __bf16* __restrict__ pkT, float* __restrict__ ksum) {
    int i = blockIdx.x * blockDim.x + threadIdx.x;
    if (i >= NCH * D_) return;
    float s = 0.f;
    for (int l = 0; l < L_; ++l) s += b2f(pkT[(size_t)i * L_ + l]);
    ksum[i] = s;
}

// exclusive prefix over chunks (per batch), transposed bf16 out
__global__ void k_scan_kv(const float* __restrict__ KtV, __bf16* __restrict__ SprevT) {
    int b = blockIdx.x;
    int tid = threadIdx.x;              // 256 threads, 64 elems each
    float acc[64];
#pragma unroll
    for (int j = 0; j < 64; ++j) acc[j] = 0.f;
    const int nc = NCH / B_;
    for (int c = 0; c < nc; ++c) {
        int ch = b * nc + c;
#pragma unroll
        for (int j = 0; j < 64; ++j) {
            int idx = tid * 64 + j;
            int d = idx / D_, e = idx % D_;
            SprevT[((size_t)ch * D_ + e) * D_ + d] = f2b(acc[j]);
            acc[j] += KtV[(size_t)ch * D_ * D_ + idx];
        }
    }
}

__global__ void k_scan_ksum(const float* __restrict__ ksum, float* __restrict__ kcum) {
    int b = blockIdx.x, d = threadIdx.x;
    float acc = 0.f;
    const int nc = NCH / B_;
    for (int c = 0; c < nc; ++c) {
        int ch = b * nc + c;
        kcum[ch * D_ + d] = acc;
        acc += ksum[ch * D_ + d];
    }
}

// chunked causal linear attention; v^T chunk staged into LDS via TDM.
__global__ void k_chunk_attn(const __bf16* __restrict__ qsb, const __bf16* __restrict__ ksb,
                             const __bf16* __restrict__ vT, const __bf16* __restrict__ SprevT,
                             const float* __restrict__ kcum, float* __restrict__ loc) {
    __shared__ __bf16 attn[L_ * L_];    // 32 KB
    __shared__ __bf16 vch[D_ * L_];     // 32 KB, filled by TDM
    __shared__ float dden[L_];
    int ch = blockIdx.x;
    int wave = threadIdx.x >> 5;
    int lane = threadIdx.x & 31;
    int r0 = ch * L_;
    int m0 = wave * 16;

#if HAVE_TDM
    if (wave == 0) {
        // Tensor DMA: flat 32KB tile pull of v^T chunk into LDS (overlaps phase 1).
        const unsigned NE = D_ * L_;                       // 16384 2-byte elements
        unsigned long long ga =
            (unsigned long long)(size_t)(vT + (size_t)ch * D_ * L_);
        unsigned int lds_off = (unsigned int)(size_t)(&vch[0]);
        v4u g0;
        g0[0] = 1u;                                        // count=1, user mode
        g0[1] = lds_off;                                   // lds_addr
        g0[2] = (unsigned int)(ga & 0xffffffffu);          // global_addr lo
        g0[3] = (unsigned int)((ga >> 32) & 0x01ffffffu) | 0x80000000u; // hi | type=2
        v8i g1 = {};
        g1[0] = (1 << 16);                                 // data_size = 2 bytes
        g1[1] = (int)((NE & 0xffffu) << 16);               // tensor_dim0[15:0]
        g1[2] = (int)(((NE >> 16) & 0xffffu) | (1u << 16));// tensor_dim0[31:16], dim1=1
        g1[3] = (int)((NE & 0xffffu) << 16);               // tile_dim0
        g1[5] = (int)NE;                                   // tensor_dim0_stride
        v4i g2 = {}, g3 = {};
#if defined(__clang_major__) && __clang_major__ >= 23
        v8i g4 = {};
        __builtin_amdgcn_tensor_load_to_lds(g0, g1, g2, g3, g4, 0);
#else
        __builtin_amdgcn_tensor_load_to_lds(g0, g1, g2, g3, 0);
#endif
    }
#else
    for (int i = threadIdx.x; i < (D_ * L_) / 8; i += 256)
        ((v8bf*)vch)[i] = ((const v8bf*)(vT + (size_t)ch * D_ * L_))[i];
#endif

    const __bf16* Aq = qsb + (size_t)(r0 + m0) * D_;

    // phase 1: masked attn strip (K-outer, A reused across 8 N-tiles)
    v8f acc[8] = {};
    for (int k = 0; k < D_; k += 32) {
        v16bf a = load_frag(Aq + k, D_);
#pragma unroll
        for (int nt = 0; nt < 8; ++nt) {
            v16bf b = load_frag(ksb + (size_t)(r0 + nt * 16) * D_ + k, D_);
            acc[nt] = wmma_bf16(a, b, acc[nt]);
        }
    }
#pragma unroll
    for (int nt = 0; nt < 8; ++nt)
#pragma unroll
        for (int i = 0; i < 8; ++i) {
            int m, n; CD_INDEX(lane, i, m, n);
            float val = ((nt * 16 + n) <= (m0 + m)) ? acc[nt][i] : 0.f;
            attn[(m0 + m) * L_ + nt * 16 + n] = f2b(val);
        }
    __syncthreads();

    // phase 1.5: den[l] = rowsum(attn) + sigma(q).kcum_prev + EPS
    if (threadIdx.x < L_) {
        int l = threadIdx.x;
        float s = 0.f;
        for (int j = 0; j < L_; ++j) s += b2f(attn[l * L_ + j]);
        for (int d = 0; d < D_; ++d)
            s += b2f(qsb[(size_t)(r0 + l) * D_ + d]) * kcum[ch * D_ + d];
        dden[l] = s + EPSF;
    }
#if HAVE_TDM
    if (wave == 0) __builtin_amdgcn_s_wait_tensorcnt(0);
#endif
    __syncthreads();

    // phase 2: num = attn @ v_chunk (LDS) + sigma(q) @ S_prev
    v8f acc2[8] = {};
    for (int k = 0; k < L_; k += 32) {
        v16bf a = load_frag(attn + m0 * L_ + k, L_);
#pragma unroll
        for (int et = 0; et < 8; ++et) {
            v16bf b = load_frag(vch + (size_t)(et * 16) * L_ + k, L_);
            acc2[et] = wmma_bf16(a, b, acc2[et]);
        }
    }
    for (int k = 0; k < D_; k += 32) {
        v16bf a = load_frag(Aq + k, D_);
#pragma unroll
        for (int et = 0; et < 8; ++et) {
            v16bf b = load_frag(SprevT + ((size_t)ch * D_ + et * 16) * D_ + k, D_);
            acc2[et] = wmma_bf16(a, b, acc2[et]);
        }
    }
#pragma unroll
    for (int et = 0; et < 8; ++et)
#pragma unroll
        for (int i = 0; i < 8; ++i) {
            int m, n; CD_INDEX(lane, i, m, n);
            loc[(size_t)(r0 + m0 + m) * D_ + et * 16 + n] = acc2[et][i] / dden[m0 + m];
        }
}

__global__ void k_combine(const float* __restrict__ memo, const float* __restrict__ loc,
                          const float* __restrict__ gate, __bf16* __restrict__ outc) {
    float g = 1.f / (1.f + __expf(-gate[0]));
    for (int i = blockIdx.x * blockDim.x + threadIdx.x; i < R_ * D_;
         i += gridDim.x * blockDim.x)
        outc[i] = f2b(g * memo[i] + (1.f - g) * loc[i]);
}

// out[r,h] = sum_d outc[r,d]*w_o[h,d]; 32x64 tiles
__global__ void k_gemm_out(const __bf16* __restrict__ outc, const __bf16* __restrict__ wob,
                           float* __restrict__ out) {
    int m0 = blockIdx.x * 32, n0 = blockIdx.y * 64;
    v8f acc[2][4] = {};
    gemm_32x64(outc + (size_t)m0 * D_, D_, wob + (size_t)n0 * D_, D_, D_, acc);
    int lane = threadIdx.x & 31;
#pragma unroll
    for (int mi = 0; mi < 2; ++mi)
#pragma unroll
        for (int ni = 0; ni < 4; ++ni)
#pragma unroll
            for (int i = 0; i < 8; ++i) {
                int m, n; CD_INDEX(lane, i, m, n);
                out[(size_t)(m0 + 16 * mi + m) * H_ + n0 + 16 * ni + n] = acc[mi][ni][i];
            }
}

// M_new[d,e] = M[d,e] + sum_r sigma(k)[r,d]*diff[r,e]; 32x64 tiles, K=8192
__global__ void k_gemm_mnew(const __bf16* __restrict__ pkT, const __bf16* __restrict__ diffT,
                            const float* __restrict__ M, float* __restrict__ Mnew) {
    int d0 = blockIdx.x * 32, e0 = blockIdx.y * 64;
    v8f acc[2][4] = {};
    for (int ch = 0; ch < NCH; ++ch)
        gemm_32x64(pkT + ((size_t)ch * D_ + d0) * L_, L_,
                   diffT + (size_t)e0 * R_ + ch * L_, R_, L_, acc);
    int lane = threadIdx.x & 31;
#pragma unroll
    for (int mi = 0; mi < 2; ++mi)
#pragma unroll
        for (int ni = 0; ni < 4; ++ni)
#pragma unroll
            for (int i = 0; i < 8; ++i) {
                int m, n; CD_INDEX(lane, i, m, n);
                int d = d0 + 16 * mi + m, e = e0 + 16 * ni + n;
                Mnew[d * D_ + e] = M[d * D_ + e] + acc[mi][ni][i];
            }
}

__global__ void k_znew(const float* __restrict__ z, const float* __restrict__ ksum,
                       float* __restrict__ zout) {
    int d = threadIdx.x;
    if (d >= D_) return;
    float s = z[d];
    for (int ch = 0; ch < NCH; ++ch) s += ksum[ch * D_ + d];
    zout[d] = s;
}

// ---------------- host launcher ----------------
extern "C" void kernel_launch(void* const* d_in, const int* in_sizes, int n_in,
                              void* d_out, int out_size, void* d_ws, size_t ws_size,
                              hipStream_t stream) {
    const float* hs   = (const float*)d_in[0];
    const float* wq   = (const float*)d_in[1];
    const float* wk   = (const float*)d_in[2];
    const float* wv   = (const float*)d_in[3];
    const float* wo   = (const float*)d_in[4];
    const float* gate = (const float*)d_in[5];
    const float* M    = (const float*)d_in[6];
    const float* z    = (const float*)d_in[7];

    float* out  = (float*)d_out;
    float* Mnew = out + (size_t)R_ * H_;
    float* znew = Mnew + D_ * D_;

    char* w = (char*)d_ws;
    auto carve = [&](size_t bytes) {
        void* p = (void*)w;
        w += (bytes + 255) & ~(size_t)255;
        return p;
    };
    __bf16* hb     = (__bf16*)carve((size_t)R_ * H_ * 2);
    __bf16* wb3    = (__bf16*)carve((size_t)3 * D_ * H_ * 2);
    __bf16* wob    = (__bf16*)carve((size_t)H_ * D_ * 2);
    __bf16* Mt     = (__bf16*)carve((size_t)D_ * D_ * 2);
    __bf16* qsb    = (__bf16*)carve((size_t)R_ * D_ * 2);
    __bf16* ksb    = (__bf16*)carve((size_t)R_ * D_ * 2);
    __bf16* pkT    = (__bf16*)carve((size_t)NCH * D_ * L_ * 2);
    float*  vf     = (float*) carve((size_t)R_ * D_ * 4);
    __bf16* vT     = (__bf16*)carve((size_t)NCH * D_ * L_ * 2);
    float*  qzden  = (float*) carve((size_t)R_ * 4);
    float*  kzden  = (float*) carve((size_t)R_ * 4);
    float*  memo   = (float*) carve((size_t)R_ * D_ * 4);
    __bf16* diffT  = (__bf16*)carve((size_t)D_ * R_ * 2);
    float*  KtV    = (float*) carve((size_t)NCH * D_ * D_ * 4);
    float*  ksum   = (float*) carve((size_t)NCH * D_ * 4);
    __bf16* SprevT = (__bf16*)carve((size_t)NCH * D_ * D_ * 2);
    float*  kcum   = (float*) carve((size_t)NCH * D_ * 4);
    float*  loc    = (float*) carve((size_t)R_ * D_ * 4);
    __bf16* outc   = (__bf16*)carve((size_t)R_ * D_ * 2);
    (void)ws_size; (void)in_sizes; (void)n_in; (void)out_size;

    k_cvt_bf16<<<2048, 256, 0, stream>>>(hs, hb, R_ * H_);
    k_cvt_bf16<<<1024, 256, 0, stream>>>(wq, wb3 + 0 * (size_t)D_ * H_, D_ * H_);
    k_cvt_bf16<<<1024, 256, 0, stream>>>(wk, wb3 + 1 * (size_t)D_ * H_, D_ * H_);
    k_cvt_bf16<<<1024, 256, 0, stream>>>(wv, wb3 + 2 * (size_t)D_ * H_, D_ * H_);
    k_cvt_bf16<<<1024, 256, 0, stream>>>(wo, wob, H_ * D_);
    k_transpose_Mt<<<64, 256, 0, stream>>>(M, Mt);

    k_gemm_qkv<<<dim3(R_ / 32, 6), 32, 0, stream>>>(hb, wb3, qsb, ksb, pkT, vf, vT);
    k_row_dots<<<R_ / 256, 256, 0, stream>>>(qsb, ksb, z, qzden, kzden);

    k_retrieve<<<dim3(R_ / 32, D_ / 64), 32, 0, stream>>>(qsb, Mt, qzden, memo);
    k_vret_diff<<<dim3(R_ / 32, D_ / 64), 32, 0, stream>>>(ksb, Mt, kzden, vf, diffT);

    k_chunk_kv<<<dim3(D_ / 32, D_ / 64, NCH), 32, 0, stream>>>(pkT, vT, KtV);
    k_chunk_ksum<<<(NCH * D_) / 256, 256, 0, stream>>>(pkT, ksum);
    k_scan_kv<<<B_, 256, 0, stream>>>(KtV, SprevT);
    k_scan_ksum<<<B_, 128, 0, stream>>>(ksum, kcum);
    k_chunk_attn<<<NCH, 256, 0, stream>>>(qsb, ksb, vT, SprevT, kcum, loc);

    k_combine<<<1024, 256, 0, stream>>>(memo, loc, gate, outc);
    k_gemm_out<<<dim3(R_ / 32, H_ / 64), 32, 0, stream>>>(outc, wob, out);

    k_gemm_mnew<<<dim3(D_ / 32, D_ / 64), 32, 0, stream>>>(pkT, diffT, M, Mnew);
    k_znew<<<1, 128, 0, stream>>>(z, ksum, znew);
}
